// ImprovedGATModelWithAttention_53824530153870
// MI455X (gfx1250) — matrix operand order repaired and checked by reading, not compiled, not run on hardware
//
#include <hip/hip_runtime.h>
#include <hip/hip_bf16.h>
#include <math.h>

typedef __attribute__((ext_vector_type(16))) __bf16 v16bf;
typedef __attribute__((ext_vector_type(8)))  float  v8f;

#define HIDF 128
#define NEG_FLT_MAX (-3.402823466e38f)

// ---------------------------------------------------------------------------
// sign-aware float atomic max (works with m initialized to -FLT_MAX)
// ---------------------------------------------------------------------------
__device__ __forceinline__ void atomicMaxF(float* addr, float val) {
    if (val >= 0.0f) {
        atomicMax((int*)addr, __float_as_int(val));
    } else {
        atomicMin((unsigned int*)addr, (unsigned int)__float_as_int(val));
    }
}

// ---------------------------------------------------------------------------
// GEMM: C[nrows x 128] (+)= A[nrows x K] @ W[K x 128] (+ bias when ADD)
// bf16 WMMA 16x16x32, f32 accumulate. K % 32 == 0.
// Block = 256 threads = 8 waves. Block computes a 64-row x 128-col tile:
// wave w owns columns [16w,16w+16) for 4 stacked 16-row tiles, so each
// B fragment is reused by 4 WMMAs. A-loads are clamped to keep EXEC all-1s
// (WMMA requirement); only the stores are guarded against row >= nrows.
// ---------------------------------------------------------------------------
template <int ADD>
__global__ void gemm128_wmma(const float* __restrict__ A,
                             const float* __restrict__ W,
                             const float* __restrict__ bias,
                             float* __restrict__ C,
                             int K, int nrows) {
    const int wave = threadIdx.x >> 5;
    const int lane = threadIdx.x & 31;
    const int half = lane >> 4;        // 0: lanes 0-15, 1: lanes 16-31
    const int l15  = lane & 15;
    const int row0 = blockIdx.x * 64;
    const int colBase = wave * 16;

    v8f acc0 = {}, acc1 = {}, acc2 = {}, acc3 = {};
    for (int kt = 0; kt < K; kt += 32) {
        // B fragment (32x16): lane half 0 -> K 0..15, half 1 -> K 16..31; col = l15
        v16bf b;
        const float* wcol = W + (size_t)(kt + half * 16) * HIDF + colBase + l15;
#pragma unroll
        for (int e = 0; e < 16; ++e) {
            b[e] = (__bf16)wcol[(size_t)e * HIDF];
        }
        // 4 A fragments (16x32 each), rows clamped so EXEC stays all-ones
#pragma unroll
        for (int t = 0; t < 4; ++t) {
            int row = row0 + t * 16 + l15;
            row = (row < nrows) ? row : (nrows - 1);
            const float* arow = A + (size_t)row * K + kt + half * 16;
            v16bf a;
#pragma unroll
            for (int e = 0; e < 8; ++e) {
                a[e]     = (__bf16)arow[e];
                a[e + 8] = (__bf16)arow[e + 8];
            }
            if (t == 0) acc0 = __builtin_amdgcn_wmma_f32_16x16x32_bf16(false, a, false, b, (short)0, acc0, false, false);
            if (t == 1) acc1 = __builtin_amdgcn_wmma_f32_16x16x32_bf16(false, a, false, b, (short)0, acc1, false, false);
            if (t == 2) acc2 = __builtin_amdgcn_wmma_f32_16x16x32_bf16(false, a, false, b, (short)0, acc2, false, false);
            if (t == 3) acc3 = __builtin_amdgcn_wmma_f32_16x16x32_bf16(false, a, false, b, (short)0, acc3, false, false);
        }
    }

    const int col = colBase + l15;
    const float bv = ADD ? bias[col] : 0.0f;   // bias only in accumulate mode
#pragma unroll
    for (int t = 0; t < 4; ++t) {
        const v8f& acc = (t == 0) ? acc0 : (t == 1) ? acc1 : (t == 2) ? acc2 : acc3;
#pragma unroll
        for (int r = 0; r < 8; ++r) {
            const int row = row0 + t * 16 + r + half * 8;  // C/D: VGPR r, lanes>=16 -> M=r+8
            if (row < nrows) {
                const size_t idx = (size_t)row * HIDF + col;
                const float v = acc[r] + bv;
                if (ADD) C[idx] += v; else C[idx] = v;
            }
        }
    }
}

// ---------------------------------------------------------------------------
// per-node attention scores: sc_s[n] = h[n,:]·a_s, sc_d[n] = h[n,:]·a_d
// one wave (32 lanes) per node
// ---------------------------------------------------------------------------
__global__ void node_scores_kernel(const float* __restrict__ h,
                                   const float* __restrict__ a_s,
                                   const float* __restrict__ a_d,
                                   float* __restrict__ sc_s,
                                   float* __restrict__ sc_d, int nrows) {
    const int node = blockIdx.x * (blockDim.x >> 5) + (threadIdx.x >> 5);
    if (node >= nrows) return;
    const int lane = threadIdx.x & 31;
    float ss = 0.0f, sd = 0.0f;
#pragma unroll
    for (int j = 0; j < 4; ++j) {
        const int c = lane + 32 * j;
        const float hv = h[(size_t)node * HIDF + c];
        ss += hv * a_s[c];
        sd += hv * a_d[c];
    }
    for (int off = 16; off > 0; off >>= 1) {
        ss += __shfl_down(ss, off, 32);
        sd += __shfl_down(sd, off, 32);
    }
    if (lane == 0) { sc_s[node] = ss; sc_d[node] = sd; }
}

// ---------------------------------------------------------------------------
// per-layer init: agg = 0 (N*128), m = -FLT_MAX, den = 0 (N)
// ---------------------------------------------------------------------------
__global__ void init_layer_kernel(float* __restrict__ agg,
                                  float* __restrict__ m,
                                  float* __restrict__ den, int N) {
    const size_t idx = (size_t)blockIdx.x * blockDim.x + threadIdx.x;
    if (idx < (size_t)N * HIDF) agg[idx] = 0.0f;
    if (idx < (size_t)N) { m[idx] = NEG_FLT_MAX; den[idx] = 0.0f; }
}

// ---------------------------------------------------------------------------
// pass 1: leaky-relu edge score + segment max into m[dst]
// ---------------------------------------------------------------------------
__global__ void edge_score_max_kernel(const int* __restrict__ ei,
                                      const float* __restrict__ sc_s,
                                      const float* __restrict__ sc_d,
                                      float* __restrict__ score,
                                      float* __restrict__ m,
                                      int E, int Etot) {
    const int e = blockIdx.x * blockDim.x + threadIdx.x;
    if (e >= Etot) return;
    const int s = (e < E) ? ei[e]     : (e - E);
    const int d = (e < E) ? ei[E + e] : (e - E);
    float sc = sc_s[s] + sc_d[d];
    sc = (sc > 0.0f) ? sc : 0.2f * sc;
    score[e] = sc;
    atomicMaxF(&m[d], sc);
}

// ---------------------------------------------------------------------------
// pass 2: e = exp(score - m[dst]); den[dst] += e
// ---------------------------------------------------------------------------
__global__ void edge_exp_kernel(const int* __restrict__ ei,
                                float* __restrict__ score,
                                const float* __restrict__ m,
                                float* __restrict__ den,
                                int E, int Etot) {
    const int e = blockIdx.x * blockDim.x + threadIdx.x;
    if (e >= Etot) return;
    const int d = (e < E) ? ei[E + e] : (e - E);
    const float ev = expf(score[e] - m[d]);
    score[e] = ev;
    atomicAdd(&den[d], ev);
}

// ---------------------------------------------------------------------------
// pass 3: alpha = e/den[dst]; agg[dst,:] += alpha * h[src,:]
// 128 threads per edge (2 edges per 256-thread block)
// ---------------------------------------------------------------------------
__global__ void edge_aggregate_kernel(const int* __restrict__ ei,
                                      const float* __restrict__ hlin,
                                      const float* __restrict__ evals,
                                      const float* __restrict__ den,
                                      float* __restrict__ agg,
                                      float* __restrict__ alpha_out,
                                      int E, int Etot) {
    const int group = threadIdx.x >> 7;
    const int c = threadIdx.x & (HIDF - 1);
    const long e = (long)blockIdx.x * 2 + group;
    if (e >= Etot) return;
    const int s = (e < E) ? ei[e]     : (int)(e - E);
    const int d = (e < E) ? ei[E + e] : (int)(e - E);
    const float alpha = evals[e] / den[d];
    if (c == 0) alpha_out[e] = alpha;
    atomicAdd(&agg[(size_t)d * HIDF + c], hlin[(size_t)s * HIDF + c] * alpha);
}

// ---------------------------------------------------------------------------
// layernorm(agg + b) [* g + be], optional elu, + optional residual
// one wave per node, 4 features per lane
// ---------------------------------------------------------------------------
__global__ void layernorm_res_kernel(const float* __restrict__ agg,
                                     const float* __restrict__ b,
                                     const float* __restrict__ g,
                                     const float* __restrict__ be,
                                     const float* __restrict__ res,
                                     float* __restrict__ out,
                                     int nrows, int do_elu) {
    const int node = blockIdx.x * (blockDim.x >> 5) + (threadIdx.x >> 5);
    if (node >= nrows) return;
    const int lane = threadIdx.x & 31;
    float v[4];
    float sum = 0.0f, sumsq = 0.0f;
#pragma unroll
    for (int j = 0; j < 4; ++j) {
        const int c = lane + 32 * j;
        v[j] = agg[(size_t)node * HIDF + c] + b[c];
        sum += v[j];
        sumsq += v[j] * v[j];
    }
    for (int off = 16; off > 0; off >>= 1) {
        sum   += __shfl_down(sum, off, 32);
        sumsq += __shfl_down(sumsq, off, 32);
    }
    sum   = __shfl(sum, 0, 32);
    sumsq = __shfl(sumsq, 0, 32);
    const float mu  = sum * (1.0f / HIDF);
    const float var = sumsq * (1.0f / HIDF) - mu * mu;
    const float rstd = rsqrtf(var + 1e-5f);
#pragma unroll
    for (int j = 0; j < 4; ++j) {
        const int c = lane + 32 * j;
        float t = (v[j] - mu) * rstd * g[c] + be[c];
        if (do_elu) t = (t > 0.0f) ? t : (expf(t) - 1.0f);
        if (res) t += res[(size_t)node * HIDF + c];
        out[(size_t)node * HIDF + c] = t;
    }
}

// ---------------------------------------------------------------------------
// pooling init + accumulate
// ---------------------------------------------------------------------------
__global__ void init_pool_kernel(float* __restrict__ psum,
                                 float* __restrict__ pmax,
                                 float* __restrict__ cnts) {
    const int idx = blockIdx.x * blockDim.x + threadIdx.x;
    if (idx < 32 * HIDF) { psum[idx] = 0.0f; pmax[idx] = NEG_FLT_MAX; }
    if (idx < 32) cnts[idx] = 0.0f;
}

__global__ void pool_kernel(const float* __restrict__ h,
                            const int* __restrict__ batch,
                            float* __restrict__ psum,
                            float* __restrict__ pmax,
                            float* __restrict__ cnts, int N) {
    const int group = threadIdx.x >> 7;
    const int c = threadIdx.x & (HIDF - 1);
    const long node = (long)blockIdx.x * 2 + group;
    if (node >= N) return;
    const int gidx = batch[node];
    const float v = h[(size_t)node * HIDF + c];
    atomicAdd(&psum[(size_t)gidx * HIDF + c], v);
    atomicMaxF(&pmax[(size_t)gidx * HIDF + c], v);
    if (c == 0) atomicAdd(&cnts[gidx], 1.0f);
}

// ---------------------------------------------------------------------------
// MLP head: one block per graph (32 graphs, tiny)
// ---------------------------------------------------------------------------
__global__ void head_kernel(const float* __restrict__ psum,
                            const float* __restrict__ pmax,
                            const float* __restrict__ cnts,
                            const float* __restrict__ c1W, const float* __restrict__ c1b,
                            const float* __restrict__ c2W, const float* __restrict__ c2b,
                            const float* __restrict__ c3W, const float* __restrict__ c3b,
                            float* __restrict__ out) {
    __shared__ float z[2 * HIDF];
    __shared__ float z2[HIDF];
    __shared__ float z3[64];
    const int g = blockIdx.x;
    const int t = threadIdx.x;   // 128 threads
    const float cnt = fmaxf(cnts[g], 1.0f);
    z[t]        = psum[(size_t)g * HIDF + t] / cnt;
    z[HIDF + t] = pmax[(size_t)g * HIDF + t];
    __syncthreads();
    float acc = c1b[t];
    for (int k = 0; k < 2 * HIDF; ++k) acc += z[k] * c1W[(size_t)k * HIDF + t];
    z2[t] = fmaxf(acc, 0.0f);
    __syncthreads();
    if (t < 64) {
        acc = c2b[t];
        for (int k = 0; k < HIDF; ++k) acc += z2[k] * c2W[(size_t)k * 64 + t];
        z3[t] = fmaxf(acc, 0.0f);
    }
    __syncthreads();
    if (t < 4) {
        acc = c3b[t];
        for (int k = 0; k < 64; ++k) acc += z3[k] * c3W[(size_t)k * 4 + t];
        out[(size_t)g * 4 + t] = acc;
    }
}

// ---------------------------------------------------------------------------
extern "C" void kernel_launch(void* const* d_in, const int* in_sizes, int n_in,
                              void* d_out, int out_size, void* d_ws, size_t ws_size,
                              hipStream_t stream) {
    const float* x     = (const float*)d_in[0];
    const int*   ei    = (const int*)d_in[1];
    const int*   batch = (const int*)d_in[2];

    const int N    = in_sizes[2];
    const int E    = in_sizes[1] / 2;
    const int FIN  = in_sizes[0] / N;
    const int Etot = E + N;

    const float *Wl[4], *As[4], *Ad[4], *Bb[4], *G[4], *Be[4];
    for (int i = 0; i < 4; ++i) {
        const int base = 3 + 6 * i;
        Wl[i] = (const float*)d_in[base + 0];
        As[i] = (const float*)d_in[base + 1];
        Ad[i] = (const float*)d_in[base + 2];
        Bb[i] = (const float*)d_in[base + 3];
        G[i]  = (const float*)d_in[base + 4];
        Be[i] = (const float*)d_in[base + 5];
    }
    const float* skipW = (const float*)d_in[27];
    const float* skipb = (const float*)d_in[28];
    const float* c1W   = (const float*)d_in[29];
    const float* c1b   = (const float*)d_in[30];
    const float* c2W   = (const float*)d_in[31];
    const float* c2b   = (const float*)d_in[32];
    const float* c3W   = (const float*)d_in[33];
    const float* c3b   = (const float*)d_in[34];

    float* ws = (float*)d_ws;
    const size_t nh = (size_t)N * HIDF;
    float* bufA  = ws;                 // lin output (per layer)
    float* bufB  = bufA + nh;
    float* bufC  = bufB + nh;
    float* sc_s  = bufC + nh;
    float* sc_d  = sc_s + N;
    float* mbuf  = sc_d + N;
    float* den   = mbuf + N;
    float* evals = den + N;
    float* psum  = evals + Etot;
    float* pmax  = psum + 32 * HIDF;
    float* cnts  = pmax + 32 * HIDF;

    float* out  = (float*)d_out;
    float* atts = out + 32 * 4;

    const int rowBlks64 = (N + 63) / 64;                // 64-row WMMA macro-tiles
    const int waveBlks  = (N + 7) / 8;                  // 8 waves per 256-thr block
    const int edgeBlks  = (Etot + 255) / 256;
    const int eAggBlks  = (Etot + 1) / 2;

    // buffer rotation: layer input / aggregation target
    const float* hin[4] = { x, bufB, bufC, bufB };
    float*       agg[4] = { bufB, bufC, bufB, bufC };

    for (int i = 0; i < 4; ++i) {
        const int K = (i == 0) ? FIN : HIDF;
        // 1) h = h_in @ W_i   (bf16 WMMA, 4 row-tiles per wave)
        gemm128_wmma<0><<<rowBlks64, 256, 0, stream>>>(hin[i], Wl[i], nullptr, bufA, K, N);
        // 2) per-node src/dst scores
        node_scores_kernel<<<waveBlks, 256, 0, stream>>>(bufA, As[i], Ad[i], sc_s, sc_d, N);
        // 3) reset agg / segment-max / segment-sum state
        init_layer_kernel<<<(int)((nh + 255) / 256), 256, 0, stream>>>(agg[i], mbuf, den, N);
        // 4) edge scores + segment max
        edge_score_max_kernel<<<edgeBlks, 256, 0, stream>>>(ei, sc_s, sc_d, evals, mbuf, E, Etot);
        // 5) exp + segment sum
        edge_exp_kernel<<<edgeBlks, 256, 0, stream>>>(ei, evals, mbuf, den, E, Etot);
        // 6) alpha + weighted scatter aggregation (+ attention output)
        edge_aggregate_kernel<<<eAggBlks, 256, 0, stream>>>(
            ei, bufA, evals, den, agg[i], atts + (size_t)i * Etot, E, Etot);
        // 7) layernorm (+bias) [+elu] [+residual]   (layer 0 residual = skip, added below)
        const float* res = (i == 0) ? nullptr : hin[i];
        layernorm_res_kernel<<<waveBlks, 256, 0, stream>>>(
            agg[i], Bb[i], G[i], Be[i], res, agg[i], N, (i < 3) ? 1 : 0);
        // 8) fused skip connection: bufB += x @ skipW + skipb
        if (i == 0) {
            gemm128_wmma<1><<<rowBlks64, 256, 0, stream>>>(x, skipW, skipb, agg[0], FIN, N);
        }
    }

    // pooling (final node features live in bufC)
    init_pool_kernel<<<(32 * HIDF + 255) / 256, 256, 0, stream>>>(psum, pmax, cnts);
    pool_kernel<<<(N + 1) / 2, 256, 0, stream>>>(bufC, batch, psum, pmax, cnts, N);
    head_kernel<<<32, 128, 0, stream>>>(psum, pmax, cnts, c1W, c1b, c2W, c2b, c3W, c3b, out);
}
